// TransformerMapper_70231305224457
// MI455X (gfx1250) — matrix operand order repaired and checked by reading, not compiled, not run on hardware
//
#include <hip/hip_runtime.h>
#include <hip/hip_bf16.h>
#include <math.h>

// ---------------------------------------------------------------------------
// Sizes (compile-time constants from the reference)
// ---------------------------------------------------------------------------
#define HEADS     8
#define DIM_HEAD  96
#define DIM       768            // HEADS*DIM_HEAD
#define DEPTH     8
#define TOPK      32
#define CLIP_LEN  10
#define PREFIX_LEN 10
#define NTOK      20             // CLIP_LEN + PREFIX_LEN
#define BATCH     16
#define ROWS      (BATCH*NTOK)   // 320
#define NKEYS     32768
#define FFDIM     3072

typedef __attribute__((ext_vector_type(16))) __bf16 v16bf;
typedef __attribute__((ext_vector_type(8)))  float  v8f;
typedef __attribute__((ext_vector_type(4)))  unsigned int u32x4;
typedef __attribute__((ext_vector_type(8)))  int    i32x8;
typedef __attribute__((ext_vector_type(4)))  int    i32x4;

union AFrg { v16bf v; __bf16 e[16]; uint4 q[2]; };
union CFrg { v8f   v; float  e[8];  };

// ---------------------------------------------------------------------------
// TDM: issue a 2-D tensor_load_to_lds of a (16 keys x 96 f32) mem_k tile.
// Descriptor bit layout per CDNA5 ISA ch.8 (group0 / group1). This toolchain
// exposes the 6-arg builtin; groups 2/3/extra zero-filled for a 2-D tensor.
// ---------------------------------------------------------------------------
__device__ __forceinline__ void tdm_load_tile(unsigned lds_off, const float* gptr)
{
    unsigned long long ga = (unsigned long long)(size_t)gptr;
    u32x4 g0;
    g0[0] = 1u;                                   // count=1, user descriptor
    g0[1] = lds_off;                              // lds_addr (bytes)
    g0[2] = (unsigned)(ga & 0xFFFFFFFFu);         // global_addr[31:0]
    g0[3] = (unsigned)((ga >> 32) & 0x01FFFFFFu)  // global_addr[56:32]
          | (2u << 30);                           // type = 2 ("image")
    i32x8 g1;
    g1[0] = 0x00020000;                           // data_size=2 (4 bytes)
    g1[1] = (int)(96u << 16);                     // tensor_dim0 = 96 (lo16)
    g1[2] = (int)0x80000000;                      // dim0 hi=0 | tensor_dim1=32768 lo16
    g1[3] = (int)(96u << 16);                     // dim1 hi=0 | tile_dim0 = 96
    g1[4] = 16;                                   // tile_dim1 = 16, tile_dim2 = 0
    g1[5] = 96;                                   // tensor_dim0_stride lo32 = 96
    g1[6] = 0;                                    // dim0_stride hi | dim1_stride lo
    g1[7] = 0;
    i32x4 gz4 = {0, 0, 0, 0};
    i32x8 gz8 = {0, 0, 0, 0, 0, 0, 0, 0};
    __builtin_amdgcn_tensor_load_to_lds(g0, g1, gz4, gz4, gz8, 0);
}

// fast tanh-approx GELU (same formula as jax.nn.gelu approximate=True)
__device__ __forceinline__ float gelu_f(float x)
{
    float u = 0.7978845608f * (x + 0.044715f * x * x * x);
    float th = 1.f - 2.f / (1.f + __expf(2.f * u));   // tanh(u)
    return 0.5f * x * (1.f + th);
}

// ---------------------------------------------------------------------------
// Generic bf16-WMMA GEMM: C[M,N] = act(alpha*(A@B) + bias) + Res
// A (M,K) f32 row-major, B (K,N) f32 row-major.
// Requires: K % 32 == 0, N % 64 == 0, M % 16 == 0.
// Block = 128 threads (4 waves), tile = 16 x 64, one 16x16 WMMA per wave.
// Tiles staged through LDS with b128 global loads; B stored transposed so
// every lane's fragment is two contiguous 16B runs (ds_load_b128 pairs).
// ---------------------------------------------------------------------------
__global__ __launch_bounds__(128)
void wmma_gemm(const float* __restrict__ A, const float* __restrict__ B,
               const float* __restrict__ bias, const float* __restrict__ Res,
               float* __restrict__ C, int M, int N, int K,
               float alpha, int gelu)
{
    __shared__ __align__(16) __bf16 As[16][40];    // [m][k], 80B rows (16B mult)
    __shared__ __align__(16) __bf16 Bst[64][40];   // [n][k], transposed

    const int tid  = threadIdx.x;
    const int lane = tid & 31;
    const int wave = tid >> 5;
    const int half = lane >> 4;            // 0 | 1
    const int l15  = lane & 15;
    const int m0   = blockIdx.y * 16;
    const int nblk = blockIdx.x * 64;
    const int nn   = nblk + wave * 16 + l15;

    v8f acc = {};
    for (int k0 = 0; k0 < K; k0 += 32) {
        // ---- A tile: 512 f32, one b128 load per thread
        {
            int idx = tid * 4, r = idx >> 5, c = idx & 31;
            float4 a4 = *(const float4*)(A + (size_t)(m0 + r) * K + k0 + c);
            As[r][c + 0] = (__bf16)a4.x; As[r][c + 1] = (__bf16)a4.y;
            As[r][c + 2] = (__bf16)a4.z; As[r][c + 3] = (__bf16)a4.w;
        }
        // ---- B tile: 2048 f32, four b128 loads per thread, stored transposed
        #pragma unroll
        for (int it = 0; it < 4; ++it) {
            int idx = (it * 128 + tid) * 4;
            int r = idx >> 6, c = idx & 63;          // r = k, c = n
            float4 b4 = *(const float4*)(B + (size_t)(k0 + r) * N + nblk + c);
            Bst[c + 0][r] = (__bf16)b4.x; Bst[c + 1][r] = (__bf16)b4.y;
            Bst[c + 2][r] = (__bf16)b4.z; Bst[c + 3][r] = (__bf16)b4.w;
        }
        __syncthreads();

        AFrg af, bf;
        // A: elem i = K[16h+i] at row m -> one contiguous 32B run
        af.q[0] = *(const uint4*)&As[l15][half * 16];
        af.q[1] = *(const uint4*)&As[l15][half * 16 + 8];
        // B: elem i = K[16h+i] at row n -> one contiguous 32B run
        bf.q[0] = *(const uint4*)&Bst[wave * 16 + l15][half * 16];
        bf.q[1] = *(const uint4*)&Bst[wave * 16 + l15][half * 16 + 8];

        acc = __builtin_amdgcn_wmma_f32_16x16x32_bf16(false, af.v, false, bf.v,
                                                      (short)0, acc, false, false);
        __syncthreads();
    }

    CFrg cf; cf.v = acc;
    #pragma unroll
    for (int r = 0; r < 8; ++r) {
        int mm = m0 + r + half * 8;
        float v = cf.e[r] * alpha;
        if (bias) v += bias[nn];
        if (gelu) v = gelu_f(v);
        if (Res) v += Res[(size_t)mm * N + nn];
        C[(size_t)mm * N + nn] = v;
    }
}

// ---------------------------------------------------------------------------
// t assembly: rows 0..9 = linear output, rows 10..19 = prefix_const
// ---------------------------------------------------------------------------
__global__ __launch_bounds__(256)
void k_assemble(const float* __restrict__ hbuf, const float* __restrict__ prefix,
                float* __restrict__ t)
{
    const int row = blockIdx.x;            // b*20 + tok
    const int b = row / NTOK, tk = row % NTOK;
    for (int c = threadIdx.x; c < DIM; c += 256) {
        float v = (tk < CLIP_LEN)
                    ? hbuf[(size_t)b * (CLIP_LEN * DIM) + tk * DIM + c]
                    : prefix[(size_t)(tk - CLIP_LEN) * DIM + c];
        t[(size_t)row * DIM + c] = v;
    }
}

// ---------------------------------------------------------------------------
// LayerNorm over 768 dims, one block per row
// ---------------------------------------------------------------------------
__device__ __forceinline__ float blk_sum768(float s, float* red)
{
    #pragma unroll
    for (int off = 16; off; off >>= 1) s += __shfl_xor(s, off, 32);
    const int lane = threadIdx.x & 31, wv = threadIdx.x >> 5;
    if (lane == 0) red[wv] = s;
    __syncthreads();
    float tot = 0.f;
    #pragma unroll
    for (int t = 0; t < 8; ++t) tot += red[t];
    __syncthreads();
    return tot;
}

__global__ __launch_bounds__(256)
void k_ln(const float* __restrict__ x, const float* __restrict__ g,
          const float* __restrict__ bb, float* __restrict__ y)
{
    __shared__ float red[8];
    const int row = blockIdx.x;
    const float* xr = x + (size_t)row * DIM;
    float v[3], s = 0.f;
    #pragma unroll
    for (int t = 0; t < 3; ++t) { v[t] = xr[threadIdx.x + t * 256]; s += v[t]; }
    const float mean = blk_sum768(s, red) * (1.f / DIM);
    float sq = 0.f;
    #pragma unroll
    for (int t = 0; t < 3; ++t) { float d = v[t] - mean; sq += d * d; }
    const float rstd = rsqrtf(blk_sum768(sq, red) * (1.f / DIM) + 1e-5f);
    #pragma unroll
    for (int t = 0; t < 3; ++t) {
        int c = threadIdx.x + t * 256;
        y[(size_t)row * DIM + c] = (v[t] - mean) * rstd * g[c] + bb[c];
    }
}

// Final LN + slice t[:,10:] -> out (16,10,768)
__global__ __launch_bounds__(256)
void k_ln_final(const float* __restrict__ x, const float* __restrict__ g,
                const float* __restrict__ bb, float* __restrict__ out)
{
    __shared__ float red[8];
    const int row = blockIdx.x;                    // 0..159
    const int b = row / CLIP_LEN, tk = row % CLIP_LEN + CLIP_LEN;
    const float* xr = x + (size_t)(b * NTOK + tk) * DIM;
    float v[3], s = 0.f;
    #pragma unroll
    for (int t = 0; t < 3; ++t) { v[t] = xr[threadIdx.x + t * 256]; s += v[t]; }
    const float mean = blk_sum768(s, red) * (1.f / DIM);
    float sq = 0.f;
    #pragma unroll
    for (int t = 0; t < 3; ++t) { float d = v[t] - mean; sq += d * d; }
    const float rstd = rsqrtf(blk_sum768(sq, red) * (1.f / DIM) + 1e-5f);
    #pragma unroll
    for (int t = 0; t < 3; ++t) {
        int c = threadIdx.x + t * 256;
        out[(size_t)row * DIM + c] = (v[t] - mean) * rstd * g[c] + bb[c];
    }
}

// ---------------------------------------------------------------------------
// Memory-attention top-k: one block per (b,h); 8 waves stream key tiles via
// the Tensor Data Mover into double-buffered LDS, score via WMMA (q is
// pre-scaled), keep per-wave running top-32 lists in LDS, merge across waves.
// ---------------------------------------------------------------------------
__global__ __launch_bounds__(256)
void k_mem_topk(const float* __restrict__ q, const float* __restrict__ memk,
                float* __restrict__ topv, int* __restrict__ topi)
{
    const int bh = blockIdx.x;             // b*8 + h
    const int b = bh >> 3, h = bh & 7;
    const int lane = threadIdx.x & 31, wave = threadIdx.x >> 5;
    const int half = lane >> 4, l15 = lane & 15;

    __shared__ __align__(16) __bf16 Qs[32][96];   // 192B rows (16B mult)
    __shared__ __align__(16) float  Ks[8][2][16 * 96]; // per-wave TDM buffers
    __shared__ float  Ss[8][20][16];
    __shared__ float  Lv[8][20][32];
    __shared__ int    Li[8][20][32];
    __shared__ float  Lmin[8][20];

    for (int idx = threadIdx.x; idx < 32 * 96; idx += 256) {
        int r = idx / 96, d = idx % 96;
        float v = (r < NTOK) ? q[(size_t)(b * NTOK + r) * DIM + h * DIM_HEAD + d] : 0.f;
        Qs[r][d] = (__bf16)v;
    }
    for (int idx = threadIdx.x; idx < 8 * 20 * 32; idx += 256) {
        (&Lv[0][0][0])[idx] = -1e30f;
        (&Li[0][0][0])[idx] = 0;
    }
    for (int idx = threadIdx.x; idx < 8 * 20; idx += 256) (&Lmin[0][0])[idx] = -1e30f;
    __syncthreads();

    const unsigned ldsA = (unsigned)(size_t)(void*)&Ks[wave][0][0];
    const unsigned ldsB = (unsigned)(size_t)(void*)&Ks[wave][1][0];
    const float* mkb = memk + (size_t)b * NKEYS * DIM_HEAD;

    const int ntiles = NKEYS >> 4;         // 2048, divisible by 8 waves
    // prime the pipeline
    tdm_load_tile(ldsA, mkb + (size_t)wave * 16 * DIM_HEAD);
    int bufi = 0;
    for (int kt = wave; kt < ntiles; kt += 8) {
        const int ktn = kt + 8;
        if (ktn < ntiles)
            tdm_load_tile(bufi ? ldsA : ldsB, mkb + (size_t)ktn * 16 * DIM_HEAD);
        // wait for the older (current) TDM transfer
        if (ktn < ntiles) __builtin_amdgcn_s_wait_tensorcnt(1);
        else              __builtin_amdgcn_s_wait_tensorcnt(0);

        const float* kb = &Ks[wave][bufi][0];   // [key][d], key-major
        v8f acc0 = {}, acc1 = {};
        #pragma unroll
        for (int ks = 0; ks < 3; ++ks) {
            const int k0 = ks * 32;
            AFrg bf, a0, a1;
            #pragma unroll
            for (int i = 0; i < 16; ++i)        // B: K(feat) = k0+8*half+i (i<8), k0+16+8*half+i-8 (i>=8), N(key)=l15
                bf.e[i] = (__bf16)kb[l15 * DIM_HEAD + k0 + (i < 8 ? half * 8 + i : 16 + half * 8 + (i - 8))];
            a0.q[0] = *(const uint4*)&Qs[l15][k0 + half * 8];
            a0.q[1] = *(const uint4*)&Qs[l15][k0 + 16 + half * 8];
            a1.q[0] = *(const uint4*)&Qs[16 + l15][k0 + half * 8];
            a1.q[1] = *(const uint4*)&Qs[16 + l15][k0 + 16 + half * 8];
            acc0 = __builtin_amdgcn_wmma_f32_16x16x32_bf16(false, a0.v, false, bf.v,
                                                           (short)0, acc0, false, false);
            acc1 = __builtin_amdgcn_wmma_f32_16x16x32_bf16(false, a1.v, false, bf.v,
                                                           (short)0, acc1, false, false);
        }
        bufi ^= 1;

        // stage scores for valid query rows (0..19)
        CFrg c0, c1; c0.v = acc0; c1.v = acc1;
        #pragma unroll
        for (int r = 0; r < 8; ++r) {
            int m = r + half * 8;
            Ss[wave][m][l15] = c0.e[r];
            int m2 = 16 + m;
            if (m2 < NTOK) Ss[wave][m2][l15] = c1.e[r];
        }
        // running top-32 per row (wave-uniform control flow)
        for (int r = 0; r < NTOK; ++r) {
            float lmin = Lmin[wave][r];
            #pragma unroll 1
            for (int j = 0; j < 16; ++j) {
                float s = Ss[wave][r][j];
                if (s > lmin) {
                    int key = kt * 16 + j;
                    float v = Lv[wave][r][lane]; int src = lane;
                    #pragma unroll
                    for (int off = 16; off; off >>= 1) {
                        float ov = __shfl_xor(v, off, 32);
                        int   os = __shfl_xor(src, off, 32);
                        if (ov < v) { v = ov; src = os; }
                    }
                    if (lane == src) { Lv[wave][r][lane] = s; Li[wave][r][lane] = key; }
                    float nv = Lv[wave][r][lane];
                    #pragma unroll
                    for (int off = 16; off; off >>= 1) {
                        float ov = __shfl_xor(nv, off, 32);
                        nv = ov < nv ? ov : nv;
                    }
                    lmin = nv;
                }
            }
            Lmin[wave][r] = lmin;
        }
    }
    __syncthreads();

    // merge 8 wave-lists (256 candidates) -> top 32 per row
    for (int r = wave; r < NTOK; r += 8) {
        float lv[8]; int li[8];
        #pragma unroll
        for (int t = 0; t < 8; ++t) { lv[t] = Lv[t][r][lane]; li[t] = Li[t][r][lane]; }
        for (int it = 0; it < TOPK; ++it) {
            float bv = lv[0]; int bs = 0;
            #pragma unroll
            for (int t = 1; t < 8; ++t) if (lv[t] > bv) { bv = lv[t]; bs = t; }
            float v = bv; int src = lane;
            #pragma unroll
            for (int off = 16; off; off >>= 1) {
                float ov = __shfl_xor(v, off, 32);
                int   os = __shfl_xor(src, off, 32);
                if (ov > v || (ov == v && os < src)) { v = ov; src = os; }
            }
            if (lane == src) {
                size_t o = ((size_t)bh * NTOK + r) * TOPK + it;
                topv[o] = bv; topi[o] = li[bs];
                lv[bs] = -1e30f;
            }
        }
    }
}

// ---------------------------------------------------------------------------
// Attention combine: causal local softmax (+ optional 32 memory logits),
// one block per (b,h), one wave per query row.
// ---------------------------------------------------------------------------
__global__ __launch_bounds__(256)
void k_attn(const float* __restrict__ q, const float* __restrict__ kv,
            const float* __restrict__ topv, const int* __restrict__ topi,
            const float* __restrict__ memv, float* __restrict__ out, int has_mem)
{
    const int bh = blockIdx.x, b = bh >> 3, h = bh & 7;
    const int lane = threadIdx.x & 31, wave = threadIdx.x >> 5;

    for (int i = wave; i < NTOK; i += 8) {
        float lj = -1e30f;
        if (lane < NTOK && lane <= i) {
            float s = 0.f;
            const float* qp = q + (size_t)(b * NTOK + i) * DIM + h * DIM_HEAD;
            const float* kp = kv + (size_t)(b * NTOK + lane) * (2 * DIM_HEAD);
            #pragma unroll 4
            for (int d = 0; d < DIM_HEAD; ++d) s += qp[d] * kp[d];
            lj = s;
        }
        float ml = -1e30f; int mi = 0;
        if (has_mem) {
            size_t o = ((size_t)bh * NTOK + i) * TOPK + lane;
            ml = topv[o]; mi = topi[o];
        }
        float m = lj > ml ? lj : ml;
        #pragma unroll
        for (int off = 16; off; off >>= 1) { float o2 = __shfl_xor(m, off, 32); m = o2 > m ? o2 : m; }
        float pl = __expf(lj - m);
        float pm = has_mem ? __expf(ml - m) : 0.f;
        float z = pl + pm;
        #pragma unroll
        for (int off = 16; off; off >>= 1) z += __shfl_xor(z, off, 32);
        const float inv = 1.f / z;

        #pragma unroll
        for (int dd = 0; dd < 3; ++dd) {
            int d = lane + dd * 32;
            float acc = 0.f;
            for (int j = 0; j <= i; ++j) {
                float p = __shfl(pl, j, 32);
                acc += p * kv[(size_t)(b * NTOK + j) * (2 * DIM_HEAD) + DIM_HEAD + d];
            }
            if (has_mem) {
                for (int t = 0; t < TOPK; ++t) {
                    float p = __shfl(pm, t, 32);
                    int  ix = __shfl(mi, t, 32);
                    acc += p * memv[((size_t)b * NKEYS + ix) * DIM_HEAD + d];
                }
            }
            out[(size_t)(b * NTOK + i) * DIM + h * DIM_HEAD + d] = acc * inv;
        }
    }
}

// ---------------------------------------------------------------------------
// Host orchestration
// ---------------------------------------------------------------------------
extern "C" void kernel_launch(void* const* d_in, const int* in_sizes, int n_in,
                              void* d_out, int out_size, void* d_ws, size_t ws_size,
                              hipStream_t stream)
{
    (void)in_sizes; (void)n_in; (void)out_size; (void)ws_size;
    const float* x        = (const float*)d_in[0];
    const float* linear_w = (const float*)d_in[1];
    const float* linear_b = (const float*)d_in[2];
    const float* prefix   = (const float*)d_in[3];
    const float* ln1_g    = (const float*)d_in[4];
    const float* ln1_b    = (const float*)d_in[5];
    const float* wq       = (const float*)d_in[6];
    const float* wkv      = (const float*)d_in[7];
    const float* wo       = (const float*)d_in[8];
    const float* ln2_g    = (const float*)d_in[9];
    const float* ln2_b    = (const float*)d_in[10];
    const float* w1       = (const float*)d_in[11];
    const float* b1       = (const float*)d_in[12];
    const float* w2       = (const float*)d_in[13];
    const float* b2       = (const float*)d_in[14];
    const float* lnf_g    = (const float*)d_in[15];
    const float* lnf_b    = (const float*)d_in[16];
    const float* mem_k    = (const float*)d_in[17];
    const float* mem_v    = (const float*)d_in[18];
    float* out = (float*)d_out;

    float* t    = (float*)d_ws;              // 320*768
    float* y    = t    + ROWS * DIM;
    float* qb   = y    + ROWS * DIM;
    float* ao   = qb   + ROWS * DIM;
    float* kvb  = ao   + ROWS * DIM;          // 320*192
    float* h1   = kvb  + ROWS * 2 * DIM_HEAD; // 320*3072
    float* hbuf = h1   + ROWS * FFDIM;        // 16*7680
    float* topv = hbuf + BATCH * CLIP_LEN * DIM;
    int*   topi = (int*)(topv + BATCH * HEADS * NTOK * TOPK);

    const float qscale = 0.10206207261596575f;  // 96^-0.5

    // input linear: (16,640)@(640,7680)+b -> hbuf
    wmma_gemm<<<dim3(CLIP_LEN * DIM / 64, 1), 128, 0, stream>>>(
        x, linear_w, linear_b, nullptr, hbuf, BATCH, CLIP_LEN * DIM, 640, 1.f, 0);
    k_assemble<<<ROWS, 256, 0, stream>>>(hbuf, prefix, t);

    for (int l = 0; l < DEPTH; ++l) {
        k_ln<<<ROWS, 256, 0, stream>>>(t, ln1_g + l * DIM, ln1_b + l * DIM, y);
        wmma_gemm<<<dim3(DIM / 64, ROWS / 16), 128, 0, stream>>>(
            y, wq + (size_t)l * DIM * DIM, nullptr, nullptr, qb,
            ROWS, DIM, DIM, qscale, 0);
        wmma_gemm<<<dim3(2 * DIM_HEAD / 64, ROWS / 16), 128, 0, stream>>>(
            y, wkv + (size_t)l * DIM * 2 * DIM_HEAD, nullptr, nullptr, kvb,
            ROWS, 2 * DIM_HEAD, DIM, 1.f, 0);
        const int has_mem = (l == 4 || l == 5);
        if (has_mem)
            k_mem_topk<<<BATCH * HEADS, 256, 0, stream>>>(qb, mem_k, topv, topi);
        k_attn<<<BATCH * HEADS, 256, 0, stream>>>(qb, kvb, topv, topi, mem_v, ao, has_mem);
        wmma_gemm<<<dim3(DIM / 64, ROWS / 16), 128, 0, stream>>>(
            ao, wo + (size_t)l * DIM * DIM, nullptr, t, t, ROWS, DIM, DIM, 1.f, 0);
        k_ln<<<ROWS, 256, 0, stream>>>(t, ln2_g + l * DIM, ln2_b + l * DIM, y);
        wmma_gemm<<<dim3(FFDIM / 64, ROWS / 16), 128, 0, stream>>>(
            y, w1 + (size_t)l * DIM * FFDIM, b1 + (size_t)l * FFDIM, nullptr, h1,
            ROWS, FFDIM, DIM, 1.f, 1);
        wmma_gemm<<<dim3(DIM / 64, ROWS / 16), 128, 0, stream>>>(
            h1, w2 + (size_t)l * FFDIM * DIM, b2 + (size_t)l * DIM, t, t,
            ROWS, DIM, FFDIM, 1.f, 0);
    }
    k_ln_final<<<BATCH * CLIP_LEN, 256, 0, stream>>>(t, lnf_g, lnf_b, out);
}
